// VQEMA_90340342104190
// MI455X (gfx1250) — matrix operand order repaired and checked by reading, not compiled, not run on hardware
//
#include <hip/hip_runtime.h>
#include <hip/hip_bf16.h>
#include <math.h>

#define BB 16
#define DD 256
#define TT 4096
#define KK 1024
#define NN (BB * TT)                  // 65536 tokens
#define QELEMS ((size_t)BB * DD * TT) // 16,777,216
#define COMMIT 0.25f

typedef __attribute__((ext_vector_type(16))) __bf16 v16bf;
typedef __attribute__((ext_vector_type(8)))  float  v8f;

// ---------------------------------------------------------------------------
// Kernel 1: convert embedding to bf16 (for WMMA) and compute ||e_k||^2 in f32
// ---------------------------------------------------------------------------
__global__ __launch_bounds__(256)
void vq_prep(const float* __restrict__ E, __bf16* __restrict__ Ebf,
             float* __restrict__ e2) {
  const int k = blockIdx.x;      // 0..K-1
  const int d = threadIdx.x;     // 0..255
  float v = E[(size_t)k * DD + d];
  Ebf[(size_t)k * DD + d] = (__bf16)v;
  __shared__ float red[256];
  red[d] = v * v;
  __syncthreads();
  for (int s = 128; s > 0; s >>= 1) {
    if (d < s) red[d] += red[d + s];
    __syncthreads();
  }
  if (d == 0) e2[k] = red[0];
}

// ---------------------------------------------------------------------------
// Kernel 2: bulk zero of encodings region + counts (grid-strided f2 stores)
// ---------------------------------------------------------------------------
__global__ __launch_bounds__(256)
void vq_zero(float2* __restrict__ enc2, unsigned* __restrict__ counts) {
  const size_t total = (size_t)NN * KK / 2;   // 33,554,432 float2
  const size_t stride = (size_t)gridDim.x * blockDim.x;
  size_t i = (size_t)blockIdx.x * blockDim.x + threadIdx.x;
  for (size_t j = i; j < total; j += stride) enc2[j] = make_float2(0.f, 0.f);
  if (i < KK) counts[i] = 0u;
}

// ---------------------------------------------------------------------------
// Kernel 3: WMMA distance GEMM + argmin.
// One wave handles 32 tokens (two 16-row M-tiles) vs all K=1024 codes:
// each B fragment is loaded once and feeds two WMMAs (acc0/acc1 alternate,
// so consecutive WMMAs never RAW on the same accumulator).
// ---------------------------------------------------------------------------
__global__ __launch_bounds__(128)
void vq_argmin_wmma(const float* __restrict__ x,      // [B, D, T]
                    const __bf16* __restrict__ Ebf,   // [K, D] bf16
                    const float* __restrict__ e2,     // [K]
                    int* __restrict__ idxOut) {       // [N]
  // A fragments staged in exact WMMA fragment order: [wave][mtile][lane][frag][16]
  __shared__ __bf16 sA[4][2][32][8][16];   // 64 KB
  const int wave = threadIdx.x >> 5;
  const int lane = threadIdx.x & 31;
  const int n0 = (blockIdx.x * 4 + wave) * 32;  // first of 32 tokens
  const int b  = n0 / TT;
  const int t0 = n0 % TT;                       // T % 32 == 0: never crosses b

  // Stage 32 tokens x 256 dims. float4 loads: lane covers 4 consecutive t.
  const size_t xbase = (size_t)b * DD * TT + t0;
  for (int j = 0; j < 64; ++j) {
    int d  = j * 4 + (lane >> 3);          // 4 d-rows per iteration
    int m0 = (lane & 7) * 4;               // 4 consecutive tokens
    float4 v4 = *(const float4*)(x + xbase + (size_t)d * TT + m0);
    int f  = d >> 5;
    int r  = d & 31;
    int hi = (r >> 4) & 1;                 // which half-wave holds this K
    int el = r & 15;                       // fragment element 0..15
    float vv[4] = {v4.x, v4.y, v4.z, v4.w};
#pragma unroll
    for (int u = 0; u < 4; ++u) {
      int m  = m0 + u;                     // token within the 32-token tile
      int ts = m >> 4;                     // which 16-row M-tile
      int mm = m & 15;
      sA[wave][ts][mm + 16 * hi][f][el] = (__bf16)vv[u];
    }
  }
  __syncthreads();

  v16bf a0[8], a1[8];
#pragma unroll
  for (int f = 0; f < 8; ++f) {
    a0[f] = *(const v16bf*)&sA[wave][0][lane][f][0];  // one 32B read per frag
    a1[f] = *(const v16bf*)&sA[wave][1][lane][f][0];
  }

  const int col = lane & 15;
  const int hi2 = (lane >> 4) & 1;
  float minv0[8], minv1[8];
  int   mini0[8], mini1[8];
#pragma unroll
  for (int j = 0; j < 8; ++j) {
    minv0[j] = 3.4e38f; mini0[j] = 0;
    minv1[j] = 3.4e38f; mini1[j] = 0;
  }

  for (int ct = 0; ct < KK / 16; ++ct) {
    // B fragment: lane's code column = ct*16+col, K offset = e + 16*hi2
    const __bf16* Brow = Ebf + (size_t)(ct * 16 + col) * DD + hi2 * 16;
    // Preload all 8 B fragments -> compiler can clause 16 b128 loads,
    // wait once, then issue 16 WMMAs back-to-back.
    v16bf bf[8];
#pragma unroll
    for (int f = 0; f < 8; ++f) bf[f] = *(const v16bf*)(Brow + f * 32);

    v8f acc0 = {0.f, 0.f, 0.f, 0.f, 0.f, 0.f, 0.f, 0.f};
    v8f acc1 = {0.f, 0.f, 0.f, 0.f, 0.f, 0.f, 0.f, 0.f};
#pragma unroll
    for (int f = 0; f < 8; ++f) {
      acc0 = __builtin_amdgcn_wmma_f32_16x16x32_bf16(
          false, a0[f], false, bf[f], (short)0, acc0, false, false);
      acc1 = __builtin_amdgcn_wmma_f32_16x16x32_bf16(
          false, a1[f], false, bf[f], (short)0, acc1, false, false);
    }
    const int   code = ct * 16 + col;
    const float e2v  = e2[code];
#pragma unroll
    for (int j = 0; j < 8; ++j) {
      float d0 = fmaf(-2.0f, acc0[j], e2v);  // ||e||^2 - 2 x.e (row const dropped)
      float d1 = fmaf(-2.0f, acc1[j], e2v);
      if (d0 < minv0[j]) { minv0[j] = d0; mini0[j] = code; }
      if (d1 < minv1[j]) { minv1[j] = d1; mini1[j] = code; }
    }
  }

  // Min+argmin across the 16 lanes of each half-wave (xor masks stay in-half).
#pragma unroll
  for (int msk = 1; msk <= 8; msk <<= 1) {
#pragma unroll
    for (int j = 0; j < 8; ++j) {
      float ov0 = __shfl_xor(minv0[j], msk, 32);
      int   oi0 = __shfl_xor(mini0[j], msk, 32);
      if (ov0 < minv0[j] || (ov0 == minv0[j] && oi0 < mini0[j])) {
        minv0[j] = ov0; mini0[j] = oi0;
      }
      float ov1 = __shfl_xor(minv1[j], msk, 32);
      int   oi1 = __shfl_xor(mini1[j], msk, 32);
      if (ov1 < minv1[j] || (ov1 == minv1[j] && oi1 < mini1[j])) {
        minv1[j] = ov1; mini1[j] = oi1;
      }
    }
  }
  if (col == 0) {
#pragma unroll
    for (int j = 0; j < 8; ++j) {
      idxOut[n0 + j + 8 * hi2]      = mini0[j];  // M-tile 0
      idxOut[n0 + 16 + j + 8 * hi2] = mini1[j];  // M-tile 1
    }
  }
}

// ---------------------------------------------------------------------------
// Kernel 4: scatter one-hot 1.0s + histogram counts
// ---------------------------------------------------------------------------
__global__ __launch_bounds__(256)
void vq_scatter(const int* __restrict__ idx, float* __restrict__ enc,
                unsigned* __restrict__ counts) {
  int n = blockIdx.x * blockDim.x + threadIdx.x;
  if (n < NN) {
    int k = idx[n];
    enc[(size_t)n * KK + k] = 1.0f;
    atomicAdd(&counts[k], 1u);
  }
}

// ---------------------------------------------------------------------------
// Kernel 5: gather quantized (straight-through) + per-block loss partials.
// Each block covers one full t-row (fixed b,d) -> coalesced in/out.
// ---------------------------------------------------------------------------
__global__ __launch_bounds__(256)
void vq_gather(const float* __restrict__ x, const float* __restrict__ E,
               const int* __restrict__ idx, float* __restrict__ q,
               float* __restrict__ partial) {
  __shared__ float red[256];
  const size_t base = (size_t)blockIdx.x * 4096;  // 4096 = T: one (b,d) row
  const size_t bd = base / TT;
  const size_t d  = bd % DD;
  const size_t b  = bd / DD;
  float acc = 0.f;
#pragma unroll
  for (int j = 0; j < 16; ++j) {
    size_t t = (size_t)j * 256 + threadIdx.x;
    size_t i = base + t;
    int k = idx[b * TT + t];
    float e  = E[(size_t)k * DD + d];
    float xv = x[i];
    q[i] = xv + (e - xv);                 // straight-through, matches ref math
    float df = e - xv;
    acc += df * df;
  }
  red[threadIdx.x] = acc;
  __syncthreads();
  for (int s = 128; s > 0; s >>= 1) {
    if (threadIdx.x < s) red[threadIdx.x] += red[threadIdx.x + s];
    __syncthreads();
  }
  if (threadIdx.x == 0) partial[blockIdx.x] = red[0];
}

// ---------------------------------------------------------------------------
// Kernel 6: finalize loss (fixed-order sum of 4096 partials) + perplexity
// ---------------------------------------------------------------------------
__global__ __launch_bounds__(1024)
void vq_finalize(const float* __restrict__ partial,
                 const unsigned* __restrict__ counts,
                 float* __restrict__ lossOut, float* __restrict__ perpOut) {
  __shared__ float red[1024];
  float a = 0.f;
  for (int j = threadIdx.x; j < 4096; j += 1024) a += partial[j];
  red[threadIdx.x] = a;
  __syncthreads();
  for (int s = 512; s > 0; s >>= 1) {
    if (threadIdx.x < s) red[threadIdx.x] += red[threadIdx.x + s];
    __syncthreads();
  }
  float lossSum = red[0];
  __syncthreads();
  float p = (float)counts[threadIdx.x] / (float)NN;
  red[threadIdx.x] = p * logf(p + 1e-10f);
  __syncthreads();
  for (int s = 512; s > 0; s >>= 1) {
    if (threadIdx.x < s) red[threadIdx.x] += red[threadIdx.x + s];
    __syncthreads();
  }
  if (threadIdx.x == 0) {
    *lossOut = COMMIT * lossSum / (float)QELEMS;
    *perpOut = expf(-red[0]);
  }
}

// ---------------------------------------------------------------------------
extern "C" void kernel_launch(void* const* d_in, const int* in_sizes, int n_in,
                              void* d_out, int out_size, void* d_ws,
                              size_t ws_size, hipStream_t stream) {
  const float* x = (const float*)d_in[0];   // [B, D, T]
  const float* E = (const float*)d_in[1];   // [K, D]

  // workspace layout
  char* ws = (char*)d_ws;
  __bf16*   Ebf     = (__bf16*)ws;                       // 512 KB
  float*    e2      = (float*)(ws + 524288);             // 4 KB
  int*      idx     = (int*)(ws + 528384);               // 256 KB
  unsigned* counts  = (unsigned*)(ws + 790528);          // 4 KB
  float*    partial = (float*)(ws + 794624);             // 16 KB

  // output layout (tuple flattened): loss | quantized_st | perplexity | enc
  float* out      = (float*)d_out;
  float* lossOut  = out;
  float* q        = out + 1;
  float* perpOut  = out + 1 + QELEMS;
  float* enc      = out + 2 + QELEMS;

  vq_prep<<<KK, 256, 0, stream>>>(E, Ebf, e2);
  vq_zero<<<2048, 256, 0, stream>>>((float2*)enc, counts);
  vq_argmin_wmma<<<NN / 128, 128, 0, stream>>>(x, Ebf, e2, idx);
  vq_scatter<<<NN / 256, 256, 0, stream>>>(idx, enc, counts);
  vq_gather<<<(int)(QELEMS / 4096), 256, 0, stream>>>(x, E, idx, q, partial);
  vq_finalize<<<1, 1024, 0, stream>>>(partial, counts, lossOut, perpOut);
}
